// PointTransformerSeg_56624848831038
// MI455X (gfx1250) — matrix operand order repaired and checked by reading, not compiled, not run on hardware
//
#include <hip/hip_runtime.h>
#include <vector>

// ---------------------------------------------------------------------------
// Types for CDNA5 WMMA
// ---------------------------------------------------------------------------
typedef __attribute__((ext_vector_type(16))) _Float16 v16h;
typedef __attribute__((ext_vector_type(8)))  _Float16 v8h;
typedef __attribute__((ext_vector_type(8)))  float    v8f;

// ---------------------------------------------------------------------------
// f32 -> f16 conversion with zero padding into (Mp x Kp) staging buffer.
// ---------------------------------------------------------------------------
__global__ void cvt_pad_f16_kernel(const float* __restrict__ src, _Float16* __restrict__ dst,
                                   int M, int K, int Mp, int Kp) {
  int i = blockIdx.x * blockDim.x + threadIdx.x;
  int total = Mp * Kp;
  if (i >= total) return;
  int k = i % Kp;
  int r = i / Kp;
  float v = (r < M && k < K) ? src[(size_t)r * K + k] : 0.0f;
  dst[i] = (_Float16)v;
}

// ---------------------------------------------------------------------------
// WMMA GEMM on padded f16 operands:
//   Y[M,N] = Xh[Mp,Kp] * Wh[Np,Kp]^T + bias[N]
// One wave computes a 16x32 output tile (two 16x16 sub-tiles sharing the A
// fragment -> two v_wmma per 32-K step). No bounds checks in the K loop:
// operands are zero-padded. All loads are contiguous 16B/32B vector loads.
// ---------------------------------------------------------------------------
__global__ __launch_bounds__(256)
void wmma_gemm_f16(const _Float16* __restrict__ Xh, const _Float16* __restrict__ Wh,
                   const float* __restrict__ bias, float* __restrict__ Y,
                   int M, int N, int Kp, int tilesN, int tiles) {
  int wave = (blockIdx.x * blockDim.x + threadIdx.x) >> 5;
  if (wave >= tiles) return;                 // wave-uniform; EXEC stays full inside
  int lane = threadIdx.x & 31;
  int tm = wave / tilesN;
  int tn = wave % tilesN;
  int l15 = lane & 15;
  int kh  = lane >> 4;

  const _Float16* arow  = Xh + (size_t)(tm * 16 + l15) * Kp;
  const _Float16* brow0 = Wh + (size_t)(tn * 32 + l15) * Kp;
  const _Float16* brow1 = brow0 + (size_t)16 * Kp;

  v8f acc0 = {};
  v8f acc1 = {};
  for (int k0 = 0; k0 < Kp; k0 += 32) {
    __builtin_prefetch(arow + k0 + 64, 0, 0);
    // A fragment: lane half kh holds K = k0+16*kh..+15 (matching B)
    // -> two contiguous 16B loads.
    v8h a0 = *(const v8h*)(arow + k0 + 16 * kh);
    v8h a1 = *(const v8h*)(arow + k0 + 16 * kh + 8);
    v16h a = __builtin_shufflevector(a0, a1, 0, 1, 2, 3, 4, 5, 6, 7,
                                     8, 9, 10, 11, 12, 13, 14, 15);
    // B fragment (contiguous k order, same as A): lane half kh holds K = k0+16*kh..+15
    // -> one contiguous 32B load per sub-tile.
    v16h b0 = *(const v16h*)(brow0 + k0 + 16 * kh);
    v16h b1 = *(const v16h*)(brow1 + k0 + 16 * kh);
    acc0 = __builtin_amdgcn_wmma_f32_16x16x32_f16(false, a, false, b0,
                                                  (short)0, acc0, false, false);
    acc1 = __builtin_amdgcn_wmma_f32_16x16x32_f16(false, a, false, b1,
                                                  (short)0, acc1, false, false);
  }
  // C/D layout: VGPR r -> M = r + 8*(lane>=16); N = lane&15
  int mbase = tm * 16 + 8 * kh;
  int n0 = tn * 32 + l15;
  int n1 = n0 + 16;
  float bv0 = (bias && n0 < N) ? bias[n0] : 0.0f;
  float bv1 = (bias && n1 < N) ? bias[n1] : 0.0f;
#pragma unroll
  for (int r = 0; r < 8; ++r) {
    int m = mbase + r;
    if (m < M) {
      if (n0 < N) Y[(size_t)m * N + n0] = acc0[r] + bv0;
      if (n1 < N) Y[(size_t)m * N + n1] = acc1[r] + bv1;
    }
  }
}

// ---------------------------------------------------------------------------
// BatchNorm (training-mode batch stats) over flattened (M, C)
// ---------------------------------------------------------------------------
__global__ void bn_stats_kernel(const float* __restrict__ x, int M, int C,
                                float* __restrict__ mean, float* __restrict__ scale) {
  int c = blockIdx.x;
  int t = threadIdx.x;
  float s = 0.f, s2 = 0.f;
  for (int m = t; m < M; m += 256) {
    float v = x[(size_t)m * C + c];
    s += v; s2 += v * v;
  }
  __shared__ float rs[256], rq[256];
  rs[t] = s; rq[t] = s2;
  __syncthreads();
  for (int off = 128; off > 0; off >>= 1) {
    if (t < off) { rs[t] += rs[t + off]; rq[t] += rq[t + off]; }
    __syncthreads();
  }
  if (t == 0) {
    float mu  = rs[0] / (float)M;
    float var = rq[0] / (float)M - mu * mu;
    mean[c]  = mu;
    scale[c] = rsqrtf(fmaxf(var, 0.f) + 1e-5f);
  }
}

__global__ void bn_apply_kernel(const float* __restrict__ x, float* __restrict__ y,
                                const float* __restrict__ mean, const float* __restrict__ scale,
                                const float* __restrict__ g, const float* __restrict__ b,
                                int total, int C, int relu) {
  int i = blockIdx.x * blockDim.x + threadIdx.x;
  if (i >= total) return;
  int c = i % C;
  float v = (x[i] - mean[c]) * scale[c] * g[c] + b[c];
  y[i] = relu ? fmaxf(v, 0.f) : v;
}

// ---------------------------------------------------------------------------
// Brute-force KNN (top-KN smallest squared distances), register insertion sort
// ---------------------------------------------------------------------------
template <int KN>
__global__ void knn_kernel(const float* __restrict__ q, const float* __restrict__ db,
                           int* __restrict__ idx, float* __restrict__ dout,
                           int B, int NQ, int ND) {
  int i = blockIdx.x * blockDim.x + threadIdx.x;
  if (i >= B * NQ) return;
  int b = i / NQ, n = i % NQ;
  const float* Q = q + ((size_t)b * NQ + n) * 3;
  const float* D = db + (size_t)b * ND * 3;
  float qx = Q[0], qy = Q[1], qz = Q[2];
  float bd[KN]; int bi[KN];
#pragma unroll
  for (int k = 0; k < KN; ++k) { bd[k] = 3.4e38f; bi[k] = 0; }
  for (int j = 0; j < ND; ++j) {
    float dx = D[j * 3 + 0] - qx;
    float dy = D[j * 3 + 1] - qy;
    float dz = D[j * 3 + 2] - qz;
    float d = dx * dx + dy * dy + dz * dz;
    if (d < bd[KN - 1]) {
      bd[KN - 1] = d; bi[KN - 1] = j;
#pragma unroll
      for (int k = KN - 1; k > 0; --k) {
        if (bd[k] < bd[k - 1]) {
          float td = bd[k]; bd[k] = bd[k - 1]; bd[k - 1] = td;
          int ti = bi[k]; bi[k] = bi[k - 1]; bi[k - 1] = ti;
        }
      }
    }
  }
  size_t o = ((size_t)b * NQ + n) * KN;
#pragma unroll
  for (int k = 0; k < KN; ++k) {
    idx[o + k] = bi[k];
    if (dout) dout[o + k] = bd[k];
  }
}

// ---------------------------------------------------------------------------
// Farthest point sampling: one block per batch, deterministic start at 0
// ---------------------------------------------------------------------------
__global__ void fps_kernel(const float* __restrict__ xyz, float* __restrict__ distws,
                           int* __restrict__ out, int N, int npoint) {
  int b = blockIdx.x, t = threadIdx.x;
  const float* P = xyz + (size_t)b * N * 3;
  float* dist = distws + (size_t)b * N;
  int* o = out + (size_t)b * npoint;
  __shared__ int far;
  __shared__ float red[256];
  __shared__ int redi[256];
  for (int i = t; i < N; i += 256) dist[i] = 1e10f;
  if (t == 0) far = 0;
  __syncthreads();
  for (int s = 0; s < npoint; ++s) {
    if (t == 0) o[s] = far;
    __syncthreads();
    float cx = P[far * 3 + 0], cy = P[far * 3 + 1], cz = P[far * 3 + 2];
    float bmax = -1.f; int bidx = 0;
    for (int i = t; i < N; i += 256) {
      float dx = P[i * 3 + 0] - cx;
      float dy = P[i * 3 + 1] - cy;
      float dz = P[i * 3 + 2] - cz;
      float d = dx * dx + dy * dy + dz * dz;
      float nd = fminf(dist[i], d);
      dist[i] = nd;
      if (nd > bmax) { bmax = nd; bidx = i; }
    }
    red[t] = bmax; redi[t] = bidx;
    __syncthreads();
    for (int off = 128; off > 0; off >>= 1) {
      if (t < off && red[t + off] > red[t]) { red[t] = red[t + off]; redi[t] = redi[t + off]; }
      __syncthreads();
    }
    if (t == 0) far = redi[0];
    __syncthreads();
  }
}

// ---------------------------------------------------------------------------
// Gathers / elementwise glue
// ---------------------------------------------------------------------------
__global__ void gather_rows_kernel(const float* __restrict__ src, const int* __restrict__ idx,
                                   float* __restrict__ out, int B, int R, int C, int Nsrc) {
  int i = blockIdx.x * blockDim.x + threadIdx.x;
  int total = B * R * C;
  if (i >= total) return;
  int c = i % C;
  int r = (i / C) % R;
  int b = i / (C * R);
  int id = idx[(size_t)b * R + r];
  out[i] = src[((size_t)b * Nsrc + id) * C + c];
}

// rel[b,n,s,k] = xyz[b, idx[b,n,s], k] - xyz[b, n, k]
__global__ void gather_sub3_kernel(const float* __restrict__ xyz, const int* __restrict__ idx,
                                   float* __restrict__ out, int B, int N, int S) {
  int i = blockIdx.x * blockDim.x + threadIdx.x;
  int total = B * N * S * 3;
  if (i >= total) return;
  int k = i % 3;
  int s = (i / 3) % S;
  int n = (i / (3 * S)) % N;
  int b = i / (3 * S * N);
  int id = idx[((size_t)b * N + n) * S + s];
  out[i] = xyz[((size_t)b * N + id) * 3 + k] - xyz[((size_t)b * N + n) * 3 + k];
}

// w[b,n,s,c] = q[b,n,c] - kg[b,n,s,c] + pe[b,n,s,c]
__global__ void qkpe_kernel(const float* __restrict__ q, const float* __restrict__ kg,
                            const float* __restrict__ pe, float* __restrict__ w,
                            int B, int N, int S, int C) {
  int i = blockIdx.x * blockDim.x + threadIdx.x;
  int total = B * N * S * C;
  if (i >= total) return;
  int c = i % C;
  int n = (i / (C * S)) % N;
  int b = i / (C * S * N);
  w[i] = q[((size_t)b * N + n) * C + c] - kg[i] + pe[i];
}

// softmax over axis=1 (the N axis) of (B,N,S,C)
__global__ void softmax_axis1_kernel(float* __restrict__ w, int B, int N, int S, int C) {
  int i = blockIdx.x * blockDim.x + threadIdx.x;
  int total = B * S * C;
  if (i >= total) return;
  int c = i % C;
  int s = (i / C) % S;
  int b = i / (C * S);
  float* base = w + (size_t)b * N * S * C + (size_t)s * C + c;
  size_t stride = (size_t)S * C;
  float m = -3.4e38f;
  for (int n = 0; n < N; ++n) m = fmaxf(m, base[n * stride]);
  float sum = 0.f;
  for (int n = 0; n < N; ++n) { float e = __expf(base[n * stride] - m); base[n * stride] = e; sum += e; }
  float inv = 1.f / sum;
  for (int n = 0; n < N; ++n) base[n * stride] *= inv;
}

// out[b,n,ch] = sum_s w[b,n,s, ch % cs] * (v[b,n,s,ch] + pe[b,n,s,ch]) + px[b,n,ch]
__global__ void attn_out_kernel(const float* __restrict__ w, const float* __restrict__ v,
                                const float* __restrict__ pe, const float* __restrict__ px,
                                float* __restrict__ out, int B, int N, int S, int C, int cs) {
  int i = blockIdx.x * blockDim.x + threadIdx.x;
  int total = B * N * C;
  if (i >= total) return;
  int ch = i % C;
  int n = (i / C) % N;
  int b = i / (C * N);
  int cc = ch % cs;
  const float* wp = w + ((size_t)b * N + n) * S * cs + cc;
  const float* vp = v + ((size_t)b * N + n) * S * C + ch;
  const float* pp = pe + ((size_t)b * N + n) * S * C + ch;
  float acc = 0.f;
  for (int s = 0; s < S; ++s) acc += wp[(size_t)s * cs] * (vp[(size_t)s * C] + pp[(size_t)s * C]);
  out[i] = acc + px[i];
}

// feat[b,i,s, 0..C-1] = px[b, idx, :]; feat[b,i,s, C..C+2] = xyz[b,idx,:] - nxyz[b,i,:]
__global__ void concat_feat_kernel(const float* __restrict__ px, const float* __restrict__ xyz,
                                   const float* __restrict__ nxyz, const int* __restrict__ idx,
                                   float* __restrict__ feat, int B, int NP, int S, int C, int Nold) {
  int F = C + 3;
  int i = blockIdx.x * blockDim.x + threadIdx.x;
  int total = B * NP * S * F;
  if (i >= total) return;
  int f = i % F;
  int s = (i / F) % S;
  int n = (i / (F * S)) % NP;
  int b = i / (F * S * NP);
  int id = idx[((size_t)b * NP + n) * S + s];
  if (f < C)
    feat[i] = px[((size_t)b * Nold + id) * C + f];
  else
    feat[i] = xyz[((size_t)b * Nold + id) * 3 + (f - C)] - nxyz[((size_t)b * NP + n) * 3 + (f - C)];
}

__global__ void maxpool_s_kernel(const float* __restrict__ x, float* __restrict__ y,
                                 int B, int NP, int S, int C) {
  int i = blockIdx.x * blockDim.x + threadIdx.x;
  int total = B * NP * C;
  if (i >= total) return;
  int c = i % C;
  int n = (i / C) % NP;
  int b = i / (C * NP);
  const float* p = x + ((size_t)b * NP + n) * S * C + c;
  float m = -3.4e38f;
  for (int s = 0; s < S; ++s) m = fmaxf(m, p[(size_t)s * C]);
  y[i] = m;
}

__global__ void interp3_kernel(const float* __restrict__ px1, const int* __restrict__ idx,
                               const float* __restrict__ dist, float* __restrict__ out,
                               int B, int NF, int CC, int NC) {
  int i = blockIdx.x * blockDim.x + threadIdx.x;
  int total = B * NF * CC;
  if (i >= total) return;
  int c = i % CC;
  int n = (i / CC) % NF;
  int b = i / (CC * NF);
  const int* id = idx + ((size_t)b * NF + n) * 3;
  const float* dd = dist + ((size_t)b * NF + n) * 3;
  float w0 = 1.f / (dd[0] + 1e-8f);
  float w1 = 1.f / (dd[1] + 1e-8f);
  float w2 = 1.f / (dd[2] + 1e-8f);
  float ws = w0 + w1 + w2;
  out[i] = (w0 * px1[((size_t)b * NC + id[0]) * CC + c] +
            w1 * px1[((size_t)b * NC + id[1]) * CC + c] +
            w2 * px1[((size_t)b * NC + id[2]) * CC + c]) / ws;
}

__global__ void add_kernel(const float* __restrict__ a, const float* __restrict__ b,
                           float* __restrict__ y, int total) {
  int i = blockIdx.x * blockDim.x + threadIdx.x;
  if (i < total) y[i] = a[i] + b[i];
}

// xyz[b,n,k] = x[b,n,k], k<3 (x is (B,N,14))
__global__ void extract_xyz_kernel(const float* __restrict__ x, float* __restrict__ xyz,
                                   int B, int N) {
  int i = blockIdx.x * blockDim.x + threadIdx.x;
  int total = B * N * 3;
  if (i >= total) return;
  int k = i % 3;
  int r = i / 3;
  xyz[i] = x[(size_t)r * 14 + k];
}

// out[b,o,n] = sum_c h[b,n,c] * W2[o,c]   (final head, (B,O,N) layout)
__global__ void head_out_kernel(const float* __restrict__ h, const float* __restrict__ W2,
                                float* __restrict__ out, int B, int N, int C, int O) {
  int i = blockIdx.x * blockDim.x + threadIdx.x;
  if (i >= B * N) return;
  int n = i % N;
  int b = i / N;
  const float* hp = h + ((size_t)b * N + n) * C;
  for (int o = 0; o < O; ++o) {
    float acc = 0.f;
    for (int c = 0; c < C; ++c) acc += hp[c] * W2[(size_t)o * C + c];
    out[((size_t)b * O + o) * N + n] = acc;
  }
}

// ---------------------------------------------------------------------------
// Host-side orchestration
// ---------------------------------------------------------------------------
struct BlockP {
  const float *Wq, *bq, *Wk, *bk, *Wv, *bv;
  const float *Wp1, *bp1, *gp, *bp, *Wp2, *bp2;
  const float *gw1, *bw1_bn, *Ww1, *bw1, *gw2, *bw2_bn, *Ww2, *bw2;
};
struct TdP { const float *W, *b, *g, *bb; };
struct TuP { const float *W1, *b1, *g1, *bb1, *W2, *b2, *g2, *bb2; };

struct WS {
  char* base; size_t off;
  float*     f(size_t n) { float* p = (float*)(base + off);     off += ((n * 4) + 255) & ~(size_t)255; return p; }
  int*       i(size_t n) { int* p = (int*)(base + off);         off += ((n * 4) + 255) & ~(size_t)255; return p; }
  _Float16*  h(size_t n) { _Float16* p = (_Float16*)(base + off); off += ((n * 2) + 255) & ~(size_t)255; return p; }
};

static inline dim3 grid1(long long total) { return dim3((unsigned)((total + 255) / 256)); }
static inline int rup(int v, int a) { return (v + a - 1) / a * a; }

// Y[M,N] = X[M,K] * W[N,K]^T + bias  via f16 staging + WMMA.
// Scratch reuse after enqueue is safe: the stream is in-order, so later
// kernels reusing the region execute only after these complete.
static void gemm(hipStream_t s, WS& ws, const float* X, const float* W, const float* bias,
                 float* Y, int M, int N, int K) {
  size_t mark = ws.off;
  int Kp = rup(K, 32), Mp = rup(M, 16), Np = rup(N, 32);
  _Float16* Xh = ws.h((size_t)Mp * Kp);
  _Float16* Wh = ws.h((size_t)Np * Kp);
  cvt_pad_f16_kernel<<<grid1((long long)Mp * Kp), 256, 0, s>>>(X, Xh, M, K, Mp, Kp);
  cvt_pad_f16_kernel<<<grid1((long long)Np * Kp), 256, 0, s>>>(W, Wh, N, K, Np, Kp);
  int tilesM = Mp / 16, tilesN = Np / 32;
  int tiles = tilesM * tilesN;
  int blocks = (tiles + 7) / 8;  // 8 waves per 256-thread block
  wmma_gemm_f16<<<dim3(blocks), dim3(256), 0, s>>>(Xh, Wh, bias, Y, M, N, Kp, tilesN, tiles);
  ws.off = mark;
}

static void bn(hipStream_t s, const float* X, float* Y, const float* g, const float* b,
               int M, int C, float* mean, float* scale, int relu) {
  bn_stats_kernel<<<dim3(C), dim3(256), 0, s>>>(X, M, C, mean, scale);
  int total = M * C;
  bn_apply_kernel<<<grid1(total), 256, 0, s>>>(X, Y, mean, scale, g, b, total, C, relu);
}

static void stage_run(hipStream_t s, const std::vector<BlockP>& blocks, float* px,
                      const float* xyz, int B, int N, int C,
                      float* mean, float* scale, WS& ws) {
  size_t mark = ws.off;
  const int S = 16;
  int M = B * N;
  int cs = C / 8;
  int* idx = ws.i((size_t)M * S);
  knn_kernel<16><<<grid1(M), 256, 0, s>>>(xyz, xyz, idx, nullptr, B, N, N);
  float* q  = ws.f((size_t)M * C);
  float* kf = ws.f((size_t)M * C);
  float* vf = ws.f((size_t)M * C);
  float* kg = ws.f((size_t)M * S * C);
  float* vg = ws.f((size_t)M * S * C);
  float* pe = ws.f((size_t)M * S * C);
  float* rel = ws.f((size_t)M * S * 3);
  float* pt  = ws.f((size_t)M * S * 3);
  float* w  = ws.f((size_t)M * S * C);
  float* w2 = ws.f((size_t)M * S * cs);
  float* w3 = ws.f((size_t)M * S * cs);
  for (size_t bi = 0; bi < blocks.size(); ++bi) {
    const BlockP& p = blocks[bi];
    gemm(s, ws, px, p.Wq, p.bq, q, M, C, C);
    gemm(s, ws, px, p.Wk, p.bk, kf, M, C, C);
    gemm(s, ws, px, p.Wv, p.bv, vf, M, C, C);
    gather_rows_kernel<<<grid1((long long)M * S * C), 256, 0, s>>>(kf, idx, kg, B, N * S, C, N);
    gather_rows_kernel<<<grid1((long long)M * S * C), 256, 0, s>>>(vf, idx, vg, B, N * S, C, N);
    gather_sub3_kernel<<<grid1((long long)M * S * 3), 256, 0, s>>>(xyz, idx, rel, B, N, S);
    gemm(s, ws, rel, p.Wp1, p.bp1, pt, M * S, 3, 3);
    bn(s, pt, pt, p.gp, p.bp, M * S, 3, mean, scale, 1);
    gemm(s, ws, pt, p.Wp2, p.bp2, pe, M * S, C, 3);
    qkpe_kernel<<<grid1((long long)M * S * C), 256, 0, s>>>(q, kg, pe, w, B, N, S, C);
    bn(s, w, w, p.gw1, p.bw1_bn, M * S, C, mean, scale, 1);
    gemm(s, ws, w, p.Ww1, p.bw1, w2, M * S, cs, C);
    bn(s, w2, w2, p.gw2, p.bw2_bn, M * S, cs, mean, scale, 1);
    gemm(s, ws, w2, p.Ww2, p.bw2, w3, M * S, cs, cs);
    softmax_axis1_kernel<<<grid1((long long)B * S * cs), 256, 0, s>>>(w3, B, N, S, cs);
    attn_out_kernel<<<grid1((long long)M * C), 256, 0, s>>>(w3, vg, pe, px, px, B, N, S, C, cs);
  }
  ws.off = mark;
}

static void tdown_run(hipStream_t s, const TdP& p, const float* px, const float* xyz,
                      int B, int Nold, int C, int NP, int O,
                      float* out, float* nxyz, float* mean, float* scale, WS& ws,
                      float* fpsdist, int* fpsidx) {
  size_t mark = ws.off;
  const int S = 16;
  fps_kernel<<<dim3(B), dim3(256), 0, s>>>(xyz, fpsdist, fpsidx, Nold, NP);
  gather_rows_kernel<<<grid1((long long)B * NP * 3), 256, 0, s>>>(xyz, fpsidx, nxyz, B, NP, 3, Nold);
  int* idx = ws.i((size_t)B * NP * S);
  knn_kernel<16><<<grid1(B * NP), 256, 0, s>>>(nxyz, xyz, idx, nullptr, B, NP, Nold);
  int F = C + 3;
  float* feat = ws.f((size_t)B * NP * S * F);
  concat_feat_kernel<<<grid1((long long)B * NP * S * F), 256, 0, s>>>(px, xyz, nxyz, idx, feat, B, NP, S, C, Nold);
  float* f2 = ws.f((size_t)B * NP * S * O);
  gemm(s, ws, feat, p.W, p.b, f2, B * NP * S, O, F);
  bn(s, f2, f2, p.g, p.bb, B * NP * S, O, mean, scale, 1);
  maxpool_s_kernel<<<grid1((long long)B * NP * O), 256, 0, s>>>(f2, out, B, NP, S, O);
  ws.off = mark;
}

static void tup_run(hipStream_t s, const TuP& p, const float* px1, const float* xyz1,
                    int NC, int CC, const float* px2, const float* xyz2, int NF, int O,
                    float* out, int B, float* mean, float* scale, WS& ws) {
  size_t mark = ws.off;
  int* idx  = ws.i((size_t)B * NF * 3);
  float* dd = ws.f((size_t)B * NF * 3);
  knn_kernel<3><<<grid1(B * NF), 256, 0, s>>>(xyz2, xyz1, idx, dd, B, NF, NC);
  float* interp = ws.f((size_t)B * NF * CC);
  interp3_kernel<<<grid1((long long)B * NF * CC), 256, 0, s>>>(px1, idx, dd, interp, B, NF, CC, NC);
  float* a  = ws.f((size_t)B * NF * O);
  float* cb = ws.f((size_t)B * NF * O);
  gemm(s, ws, px2, p.W1, p.b1, a, B * NF, O, O);
  bn(s, a, a, p.g1, p.bb1, B * NF, O, mean, scale, 1);
  gemm(s, ws, interp, p.W2, p.b2, cb, B * NF, O, CC);
  bn(s, cb, cb, p.g2, p.bb2, B * NF, O, mean, scale, 1);
  add_kernel<<<grid1((long long)B * NF * O), 256, 0, s>>>(a, cb, out, B * NF * O);
  ws.off = mark;
}

extern "C" void kernel_launch(void* const* d_in, const int* in_sizes, int n_in,
                              void* d_out, int out_size, void* d_ws, size_t ws_size,
                              hipStream_t stream) {
  (void)in_sizes; (void)n_in; (void)out_size; (void)ws_size;
  const int B = 2, N0 = 4096;
  hipStream_t s = stream;

  // --- Parameter walk: d_in[0] = x, then params in dict-insertion order ---
  const float* x = (const float*)d_in[0];
  int cur = 1;
  auto nf = [&]() { return (const float*)d_in[cur++]; };
  auto loadBlock = [&]() {
    BlockP p;
    p.Wq = nf(); p.bq = nf(); p.Wk = nf(); p.bk = nf(); p.Wv = nf(); p.bv = nf();
    p.Wp1 = nf(); p.bp1 = nf(); p.gp = nf(); p.bp = nf(); p.Wp2 = nf(); p.bp2 = nf();
    p.gw1 = nf(); p.bw1_bn = nf(); p.Ww1 = nf(); p.bw1 = nf();
    p.gw2 = nf(); p.bw2_bn = nf(); p.Ww2 = nf(); p.bw2 = nf();
    return p;
  };
  auto loadBlocks = [&](int n) { std::vector<BlockP> v; for (int i = 0; i < n; ++i) v.push_back(loadBlock()); return v; };
  auto loadTd = [&]() { TdP t; t.W = nf(); t.b = nf(); t.g = nf(); t.bb = nf(); return t; };
  auto loadTu = [&]() { TuP t; t.W1 = nf(); t.b1 = nf(); t.g1 = nf(); t.bb1 = nf();
                        t.W2 = nf(); t.b2 = nf(); t.g2 = nf(); t.bb2 = nf(); return t; };

  const float* Win = nf(); const float* gin = nf(); const float* bin = nf();
  std::vector<BlockP> tf1 = loadBlocks(2);
  TdP td1 = loadTd(); std::vector<BlockP> tf2 = loadBlocks(3);
  TdP td2 = loadTd(); std::vector<BlockP> tf3 = loadBlocks(4);
  TdP td3 = loadTd(); std::vector<BlockP> tf4 = loadBlocks(3);
  TdP td4 = loadTd(); std::vector<BlockP> tf5 = loadBlocks(2);
  TuP tu1 = loadTu(); std::vector<BlockP> tf6 = loadBlocks(2);
  TuP tu2 = loadTu(); std::vector<BlockP> tf7 = loadBlocks(2);
  TuP tu3 = loadTu(); std::vector<BlockP> tf8 = loadBlocks(2);
  TuP tu4 = loadTu(); std::vector<BlockP> tf9 = loadBlocks(2);
  const float* Wout1 = nf(); const float* gout = nf(); const float* bout = nf();
  const float* Wout2 = nf();

  // --- Workspace: persistent buffers first, transient reset per phase ---
  WS ws{(char*)d_ws, 0};
  float* mean  = ws.f(512);
  float* scale = ws.f(512);
  float* xyz0 = ws.f((size_t)B * 4096 * 3);  float* out0 = ws.f((size_t)B * 4096 * 32);
  float* xyz1 = ws.f((size_t)B * 1024 * 3);  float* out1 = ws.f((size_t)B * 1024 * 64);
  float* xyz2 = ws.f((size_t)B * 512 * 3);   float* out2 = ws.f((size_t)B * 512 * 128);
  float* xyz3 = ws.f((size_t)B * 256 * 3);   float* out3 = ws.f((size_t)B * 256 * 256);
  float* xyz4 = ws.f((size_t)B * 128 * 3);   float* out4 = ws.f((size_t)B * 128 * 512);
  float* up1 = ws.f((size_t)B * 256 * 256);
  float* up2 = ws.f((size_t)B * 512 * 128);
  float* up3 = ws.f((size_t)B * 1024 * 64);
  float* up4 = ws.f((size_t)B * 4096 * 32);
  float* fpsdist = ws.f((size_t)B * 4096);
  int*   fpsidx  = ws.i((size_t)B * 1024);
  float* hbuf = ws.f((size_t)B * 4096 * 32);

  // --- Input embedding ---
  extract_xyz_kernel<<<grid1((long long)B * N0 * 3), 256, 0, s>>>(x, xyz0, B, N0);
  gemm(s, ws, x, Win, nullptr, out0, B * N0, 32, 14);
  bn(s, out0, out0, gin, bin, B * N0, 32, mean, scale, 1);

  // --- Encoder ---
  stage_run(s, tf1, out0, xyz0, B, 4096, 32, mean, scale, ws);
  tdown_run(s, td1, out0, xyz0, B, 4096, 32, 1024, 64, out1, xyz1, mean, scale, ws, fpsdist, fpsidx);
  stage_run(s, tf2, out1, xyz1, B, 1024, 64, mean, scale, ws);
  tdown_run(s, td2, out1, xyz1, B, 1024, 64, 512, 128, out2, xyz2, mean, scale, ws, fpsdist, fpsidx);
  stage_run(s, tf3, out2, xyz2, B, 512, 128, mean, scale, ws);
  tdown_run(s, td3, out2, xyz2, B, 512, 128, 256, 256, out3, xyz3, mean, scale, ws, fpsdist, fpsidx);
  stage_run(s, tf4, out3, xyz3, B, 256, 256, mean, scale, ws);
  tdown_run(s, td4, out3, xyz3, B, 256, 256, 128, 512, out4, xyz4, mean, scale, ws, fpsdist, fpsidx);
  stage_run(s, tf5, out4, xyz4, B, 128, 512, mean, scale, ws);

  // --- Decoder ---
  tup_run(s, tu1, out4, xyz4, 128, 512, out3, xyz3, 256, 256, up1, B, mean, scale, ws);
  stage_run(s, tf6, up1, xyz3, B, 256, 256, mean, scale, ws);
  tup_run(s, tu2, up1, xyz3, 256, 256, out2, xyz2, 512, 128, up2, B, mean, scale, ws);
  stage_run(s, tf7, up2, xyz2, B, 512, 128, mean, scale, ws);
  tup_run(s, tu3, up2, xyz2, 512, 128, out1, xyz1, 1024, 64, up3, B, mean, scale, ws);
  stage_run(s, tf8, up3, xyz1, B, 1024, 64, mean, scale, ws);
  tup_run(s, tu4, up3, xyz1, 1024, 64, out0, xyz0, 4096, 32, up4, B, mean, scale, ws);
  stage_run(s, tf9, up4, xyz0, B, 4096, 32, mean, scale, ws);

  // --- Head ---
  gemm(s, ws, up4, Wout1, nullptr, hbuf, B * N0, 32, 32);
  bn(s, hbuf, hbuf, gout, bout, B * N0, 32, mean, scale, 1);
  head_out_kernel<<<grid1((long long)B * N0), 256, 0, s>>>(hbuf, Wout2, (float*)d_out, B, N0, 32, 2);
}
